// SinkhornAttention_48747878809988
// MI455X (gfx1250) — matrix operand order repaired and checked by reading, not compiled, not run on hardware
//
#include <hip/hip_runtime.h>
#include <math.h>

typedef __attribute__((ext_vector_type(16))) _Float16 v16h;
typedef __attribute__((ext_vector_type(8)))  float    v8f;

#define BH   32
#define NB   32          // buckets
#define BS   128         // bucket size
#define DH   64
#define TLEN 4096
#define VTS  264         // padded V^T row stride (f16 elems)
#define SCALE 0.125f     // 64^-0.5

// ---------------- kernel 1: bucket means ----------------
__global__ __launch_bounds__(64)
void means_kernel(const float* __restrict__ q, const float* __restrict__ k,
                  float* __restrict__ sq, float* __restrict__ sk) {
  const int u = blockIdx.x, bh = blockIdx.y, d = threadIdx.x;
  const float* qp = q + (size_t)(bh*TLEN + u*BS)*DH + d;
  const float* kp = k + (size_t)(bh*TLEN + u*BS)*DH + d;
  float aq = 0.f, ak = 0.f;
  for (int r = 0; r < BS; ++r) { aq += qp[r*DH]; ak += kp[r*DH]; }
  sq[(bh*NB + u)*DH + d] = aq * (1.0f/BS);
  sk[(bh*NB + u)*DH + d] = ak * (1.0f/BS);
}

// ---------------- kernel 2: routing (softmax top-1) ----------------
__global__ __launch_bounds__(256)
void route_kernel(const float* __restrict__ sq, const float* __restrict__ sk,
                  float* __restrict__ rval, int* __restrict__ ridx) {
  __shared__ float R[NB*NB];
  const int bh = blockIdx.x, tid = threadIdx.x;
  for (int p = tid; p < NB*NB; p += 256) {
    const int i = p >> 5, j = p & 31;
    const float* a = sq + (bh*NB + i)*DH;
    const float* b = sk + (bh*NB + j)*DH;
    float s = 0.f;
    for (int d = 0; d < DH; ++d) s += a[d]*b[d];
    R[p] = s * SCALE;
  }
  __syncthreads();
  if (tid < NB) {
    float m = -1e30f; int am = 0;
    for (int j = 0; j < NB; ++j) { float v = R[tid*NB + j]; if (v > m) { m = v; am = j; } }
    float sum = 0.f;
    for (int j = 0; j < NB; ++j) sum += __expf(R[tid*NB + j] - m);
    ridx[bh*NB + tid] = am;
    rval[bh*NB + tid] = 1.0f / sum;   // max of softmax row == 1/sum(exp(s-m))
  }
}

// ---------------- kernel 3: bucketed attention ----------------
// dynamic LDS partition (bytes)
#define SM_S   0                       // 128x256 f32  = 131072
#define SM_P   131072                  // 128x256 f16  =  65536
#define SM_K   196608                  // 256x64  f16  =  32768
#define SM_VT  229376                  // 64xVTS  f16  =  33792
#define SMEM_MAIN (SM_VT + DH*VTS*2)   // 263168 bytes (< 320KB WGP LDS)

__global__ __launch_bounds__(256)
void attn_kernel(const float* __restrict__ q, const float* __restrict__ kin,
                 const float* __restrict__ vin,
                 const float* __restrict__ rval, const int* __restrict__ ridx,
                 float* __restrict__ out) {
  extern __shared__ char smem[];
  float*    S  = (float*)   (smem + SM_S);
  _Float16* P  = (_Float16*)(smem + SM_P);
  _Float16* KS = (_Float16*)(smem + SM_K);
  _Float16* VT = (_Float16*)(smem + SM_VT);

  const int u = blockIdx.x, bh = blockIdx.y;
  const int tid  = threadIdx.x;
  const int lane = tid & 31;
  const int w    = tid >> 5;                // wave id: owns query rows w*16..w*16+15
  const int l15  = lane & 15;
  const int hi8  = (lane < 16) ? 0 : 8;     // C-layout row offset
  const int lhalf = (lane < 16) ? 0 : 8;    // A-layout K offset
  const int koff  = (lane < 16) ? 0 : 16;   // unused; B loads also use lhalf

  const int   r_idx = ridx[bh*NB + u];
  const float r_val = rval[bh*NB + u];

  // ---- stage K_cat [256][64] and V_cat^T [64][VTS] as f16 ----
  // 8 threads per key row (32B each) -> fully coalesced 256B global reads per group
  {
    const int g   = tid >> 3;   // 32 row-groups
    const int sub = tid & 7;    // 8 cols of 8 floats
    for (int it = 0; it < 8; ++it) {
      const int r    = it*32 + g;               // key row 0..255
      const int srcB = (r < BS) ? r_idx : u;
      const float sc = (r < BS) ? r_val : 1.0f;
      const int rr   = (r < BS) ? r : r - BS;
      const float* kp = kin + (size_t)(bh*TLEN + srcB*BS + rr)*DH + sub*8;
      const float* vp = vin + (size_t)(bh*TLEN + srcB*BS + rr)*DH + sub*8;
      _Float16 kv[8], vv[8];
      for (int i = 0; i < 8; ++i) { kv[i] = (_Float16)(kp[i]*sc);
                                    vv[i] = (_Float16)(vp[i]*sc); }
      for (int i = 0; i < 8; ++i) KS[r*DH + sub*8 + i] = kv[i];
      for (int i = 0; i < 8; ++i) VT[(sub*8 + i)*VTS + r] = vv[i];
    }
  }

  // ---- load Q A-fragments (2 K-chunks of 32) straight from global, f32->f16 ----
  v16h aq0, aq1;
  {
    const float* qp = q + (size_t)(bh*TLEN + u*BS + w*16 + l15)*DH;
    for (int i = 0; i < 8; ++i) { aq0[i]   = (_Float16)qp[lhalf + i];
                                  aq0[8+i] = (_Float16)qp[lhalf + 16 + i]; }
    for (int i = 0; i < 8; ++i) { aq1[i]   = (_Float16)qp[32 + lhalf + i];
                                  aq1[8+i] = (_Float16)qp[32 + lhalf + 16 + i]; }
  }

  __syncthreads();

  // ---- S = Q * K_cat^T : 16 N-tiles x (K=64 as 2 chunks) ----
  for (int nt = 0; nt < 16; ++nt) {
    v8f c = {};
    const _Float16* kp = KS + (nt*16 + l15)*DH;   // lane -> key row = B column
    v16h b;
    for (int i = 0; i < 8; ++i) { b[i] = kp[lhalf + i]; b[8+i] = kp[lhalf + 16 + i]; }
    c = __builtin_amdgcn_wmma_f32_16x16x32_f16(false, aq0, false, b, (short)0, c, false, false);
    for (int i = 0; i < 8; ++i) { b[i] = kp[32 + lhalf + i]; b[8+i] = kp[32 + lhalf + 16 + i]; }
    c = __builtin_amdgcn_wmma_f32_16x16x32_f16(false, aq1, false, b, (short)0, c, false, false);
    float* srow = S + (w*16 + hi8)*256 + nt*16 + l15;
    for (int r = 0; r < 8; ++r) srow[r*256] = c[r];
  }

  __syncthreads();

  // ---- row softmax (scaled): 2 lanes per row, shfl_xor combine ----
  {
    const int row  = tid >> 1;        // 128 rows, all 256 threads active
    const int half = tid & 1;
    float* rp = S + row*256 + half*128;
    float m = -1e30f;
    for (int j = 0; j < 128; ++j) m = fmaxf(m, rp[j]);
    m = fmaxf(m, __shfl_xor(m, 1));
    float sum = 0.f;
    for (int j = 0; j < 128; ++j) { float e = __expf((rp[j] - m)*SCALE); rp[j] = e; sum += e; }
    sum += __shfl_xor(sum, 1);
    const float inv = 1.0f / sum;
    _Float16* pp = P + row*256 + half*128;
    for (int j = 0; j < 128; ++j) pp[j] = (_Float16)(rp[j]*inv);
  }

  __syncthreads();

  // ---- O = P * V_cat : 4 d-tiles x (K=256 as 8 chunks) ----
  const _Float16* pp = P + (w*16 + l15)*256;
  for (int nt = 0; nt < 4; ++nt) {
    v8f c = {};
    const _Float16* vp = VT + (nt*16 + l15)*VTS;  // lane -> d column = row of V^T
    for (int kc = 0; kc < 8; ++kc) {
      const int k0 = kc*32;
      v16h a, b;
      for (int i = 0; i < 8; ++i) { a[i]   = pp[k0 + lhalf + i];
                                    a[8+i] = pp[k0 + lhalf + 16 + i]; }
      for (int i = 0; i < 8; ++i) { b[i] = vp[k0 + lhalf + i]; b[8+i] = vp[k0 + lhalf + 16 + i]; }
      c = __builtin_amdgcn_wmma_f32_16x16x32_f16(false, a, false, b, (short)0, c, false, false);
    }
    float* op = out + (size_t)(bh*TLEN + u*BS + w*16 + hi8)*DH + nt*16 + l15;
    for (int r = 0; r < 8; ++r) op[r*DH] = c[r];
  }
}

// ---------------- host launcher ----------------
extern "C" void kernel_launch(void* const* d_in, const int* in_sizes, int n_in,
                              void* d_out, int out_size, void* d_ws, size_t ws_size,
                              hipStream_t stream) {
  (void)in_sizes; (void)n_in; (void)out_size; (void)ws_size;
  const float* q = (const float*)d_in[0];
  const float* k = (const float*)d_in[1];
  const float* v = (const float*)d_in[2];
  float* out = (float*)d_out;

  // workspace layout (floats): sq | sk | rval | ridx(int)
  float* sq = (float*)d_ws;
  float* sk = sq + BH*NB*DH;
  float* rv = sk + BH*NB*DH;
  int*   ri = (int*)(rv + BH*NB);

  means_kernel<<<dim3(NB, BH), 64, 0, stream>>>(q, k, sq, sk);
  route_kernel<<<BH, 256, 0, stream>>>(sq, sk, rv, ri);

  (void)hipFuncSetAttribute(reinterpret_cast<const void*>(attn_kernel),
                            hipFuncAttributeMaxDynamicSharedMemorySize, SMEM_MAIN);
  attn_kernel<<<dim3(NB, BH), 256, SMEM_MAIN, stream>>>(q, k, v, rv, ri, out);
}
